// ScaledDotProductAttentionBlock_22454089023658
// MI455X (gfx1250) — matrix operand order repaired and checked by reading, not compiled, not run on hardware
//
#include <hip/hip_runtime.h>

// Problem dims (compile-time constants from the reference)
#define NB 4
#define LL 2048
#define DD 1024
#define HH 1024
#define OO 1024

typedef __bf16 bf16_t;
typedef bf16_t v16bf __attribute__((ext_vector_type(16)));
typedef float  v8f   __attribute__((ext_vector_type(8)));

union BF16Frag { int4 q[2]; v16bf v; };

__device__ __forceinline__ unsigned short f2bf(float f) {
  union { float f; unsigned u; } c; c.f = f;
  unsigned r = c.u + 0x7FFFu + ((c.u >> 16) & 1u);   // round-to-nearest-even-ish
  return (unsigned short)(r >> 16);
}
__device__ __forceinline__ float bf2f(unsigned short h) {
  union { unsigned u; float f; } c; c.u = ((unsigned)h) << 16;
  return c.f;
}

// A-matrix 16x32 bf16 fragment: lane holds row (lane&15); K chunk
// [half*16, half*16+16) contiguous -> two 16B loads.
__device__ __forceinline__ v16bf load_a_frag(const unsigned short* rowk, int half) {
  BF16Frag u;
  u.q[0] = *reinterpret_cast<const int4*>(rowk + half * 16);
  u.q[1] = *reinterpret_cast<const int4*>(rowk + half * 16 + 8);
  return u.v;
}
// B-matrix 32x16 bf16 fragment: lane holds column (lane&15); K chunk
// [half*16, half*16+16) contiguous -> two 16B loads. caller pre-adds half*16.
__device__ __forceinline__ v16bf load_b_frag(const unsigned short* __restrict__ colk) {
  BF16Frag u;
  u.q[0] = *reinterpret_cast<const int4*>(colk);
  u.q[1] = *reinterpret_cast<const int4*>(colk + 8);
  return u.v;
}
__device__ __forceinline__ v8f wmma_bf16(v16bf a, v16bf b, v8f c) {
  return __builtin_amdgcn_wmma_f32_16x16x32_bf16(false, a, false, b, (short)0, c,
                                                 false, false);
}

// batch-load 8 column-tile B fragments for one k-step (one 16-load clause)
__device__ __forceinline__ void load_b8(v16bf b[8],
                                        const unsigned short* __restrict__ base,
                                        size_t colstride, int kofs) {
  #pragma unroll
  for (int j = 0; j < 8; ++j)
    b[j] = load_b_frag(base + (size_t)j * colstride + kofs);
}
__device__ __forceinline__ void wmma8(v8f acc[8], v16bf a, const v16bf b[8]) {
  #pragma unroll
  for (int j = 0; j < 8; ++j) acc[j] = wmma_bf16(a, b[j], acc[j]);
}

// ---- CDNA5 async memory->LDS staging (ASYNCcnt path), with safe fallback ----
#if defined(__HIP_DEVICE_COMPILE__) && \
    __has_builtin(__builtin_amdgcn_global_load_async_to_lds_b128)
#define HAVE_ASYNC_LDS 1
#else
#define HAVE_ASYNC_LDS 0
#endif

typedef int v4i_vec __attribute__((vector_size(16)));
typedef __attribute__((address_space(1))) v4i_vec gas_v4i;   // global
typedef __attribute__((address_space(3))) v4i_vec sas_v4i;   // LDS

__device__ __forceinline__ void stage16(const unsigned short* __restrict__ g,
                                        unsigned short* l) {
#if HAVE_ASYNC_LDS
  __builtin_amdgcn_global_load_async_to_lds_b128(
      (gas_v4i*)(uintptr_t)g,
      (sas_v4i*)(uintptr_t)l,   // low 32 bits of generic ptr == LDS byte address
      0, 0);
#else
  *reinterpret_cast<int4*>(l) = *reinterpret_cast<const int4*>(g);
#endif
}
__device__ __forceinline__ void stage_wait_barrier() {
#if HAVE_ASYNC_LDS
#if __has_builtin(__builtin_amdgcn_s_wait_asynccnt)
  __builtin_amdgcn_s_wait_asynccnt(0);
#else
  asm volatile("s_wait_asynccnt 0x0" ::: "memory");
#endif
#endif
  __syncthreads();
}

// ---------------- Kernel 1: x (f32) -> bf16, row-major ----------------
__global__ void cvt_x_kernel(const float* __restrict__ x,
                             unsigned short* __restrict__ xb, int npairs) {
  int i = blockIdx.x * blockDim.x + threadIdx.x;
  if (i < npairs) {
    float2 f = reinterpret_cast<const float2*>(x)[i];
    unsigned out = (unsigned)f2bf(f.x) | ((unsigned)f2bf(f.y) << 16);
    reinterpret_cast<unsigned*>(xb)[i] = out;
  }
}

// ---------------- Kernel 2: W[d][h] (f32) -> WT[h][d] (bf16) ----------------
__global__ void cvt_wT_kernel(const float* __restrict__ W,
                              unsigned short* __restrict__ WT) {
  int j = blockIdx.x * blockDim.x + threadIdx.x;  // j = h*DD + d
  int h = j >> 10;            // / DD
  int d = j & (DD - 1);
  WT[j] = f2bf(W[d * HH + h]);
}

// ---------------- Kernel 3: q/k/v projections ----------------
#define XS (DD + 8)   // padded LDS row stride (bank de-conflict)

// One 16-row x 128-col projection tile: 8 live accumulators, batched +
// double-buffered B fragments so 16 loads are in flight behind 8 WMMAs.
__device__ __forceinline__ void proj_tile(const unsigned short* xs_lds,
                                          const unsigned short* __restrict__ WT,
                                          int w, int half, int l16, v8f acc[8]) {
  const v8f zero = {};
  #pragma unroll
  for (int j = 0; j < 8; ++j) acc[j] = zero;
  const unsigned short* arow = xs_lds + l16 * XS;
  const unsigned short* wbase =
      WT + (size_t)((w * 8) * 16 + l16) * DD + half * 16;
  const size_t cstride = (size_t)16 * DD;

  v16bf bA[8], bB[8];
  load_b8(bA, wbase, cstride, 0);
  int ks = 0;
  #pragma unroll 1
  for (; ks < DD / 32 - 2; ks += 2) {
    load_b8(bB, wbase, cstride, (ks + 1) * 32);
    v16bf a = load_a_frag(arow + ks * 32, half);
    wmma8(acc, a, bA);
    load_b8(bA, wbase, cstride, (ks + 2) * 32);
    a = load_a_frag(arow + (ks + 1) * 32, half);
    wmma8(acc, a, bB);
  }
  load_b8(bB, wbase, cstride, (ks + 1) * 32);
  v16bf a = load_a_frag(arow + ks * 32, half);
  wmma8(acc, a, bA);
  a = load_a_frag(arow + (ks + 1) * 32, half);
  wmma8(acc, a, bB);
}

__global__ __launch_bounds__(256) void qkv_kernel(
    const unsigned short* __restrict__ xb,
    const unsigned short* __restrict__ WqT, const float* __restrict__ bq,
    const unsigned short* __restrict__ WkT, const float* __restrict__ bk,
    const unsigned short* __restrict__ WvT, const float* __restrict__ bv,
    unsigned short* __restrict__ qbuf, unsigned short* __restrict__ kbuf,
    unsigned short* __restrict__ vT) {
  __shared__ unsigned short xs[16 * XS];

  const int tid = threadIdx.x;
  const int rt  = blockIdx.x;                 // row tile over flat [N*L]
  const unsigned short* xrow = xb + (size_t)rt * 16 * DD;
  for (int c = tid; c < 16 * (DD / 8); c += 256) {
    int row = c >> 7, col8 = c & 127;
    stage16(xrow + row * DD + col8 * 8, &xs[row * XS + col8 * 8]);
  }
  stage_wait_barrier();

  const int w = tid >> 5, lane = tid & 31, half = lane >> 4, l16 = lane & 15;
  v8f acc[8];

  // ---- q ----
  proj_tile(xs, WqT, w, half, l16, acc);
  #pragma unroll
  for (int j = 0; j < 8; ++j) {
    const int col = (w * 8 + j) * 16 + l16;
    const float bias = bq[col];
    #pragma unroll
    for (int r = 0; r < 8; ++r) {
      int g = rt * 16 + r + half * 8;
      qbuf[(size_t)g * HH + col] = f2bf(acc[j][r] + bias);
    }
  }
  // ---- k ----
  proj_tile(xs, WkT, w, half, l16, acc);
  #pragma unroll
  for (int j = 0; j < 8; ++j) {
    const int col = (w * 8 + j) * 16 + l16;
    const float bias = bk[col];
    #pragma unroll
    for (int r = 0; r < 8; ++r) {
      int g = rt * 16 + r + half * 8;
      kbuf[(size_t)g * HH + col] = f2bf(acc[j][r] + bias);
    }
  }
  // ---- v (stored transposed: vT[n][o][m], m-contiguous) ----
  proj_tile(xs, WvT, w, half, l16, acc);
  #pragma unroll
  for (int j = 0; j < 8; ++j) {
    const int col = (w * 8 + j) * 16 + l16;
    const float bias = bv[col];
    #pragma unroll
    for (int r = 0; r < 8; ++r) {
      int g = rt * 16 + r + half * 8;
      int n = g >> 11, m = g & (LL - 1);
      vT[((size_t)n * OO + col) * LL + m] = f2bf(acc[j][r] + bias);
    }
  }
}

// ---------------- Kernel 4: attention (scores -> softmax -> P@V) ----------------
#define QS (HH + 8)
#define SS (LL + 8)

__global__ __launch_bounds__(256) void attn_kernel(
    const unsigned short* __restrict__ qbuf,
    const unsigned short* __restrict__ kbuf,
    const unsigned short* __restrict__ vT,
    float* __restrict__ out) {
  __shared__ unsigned short qs[16 * QS];   // 33,024 B
  __shared__ unsigned short sc[16 * SS];   // 65,792 B
  __shared__ float red[16 * 16];
  __shared__ float rowmax[16];
  __shared__ float rowsum[16];

  const int tid = threadIdx.x;
  const int n = blockIdx.x >> 7;           // 128 row tiles per batch
  const int rtile = blockIdx.x & 127;
  const int m0 = rtile * 16;

  // stage 16 q rows (async DMA to LDS when available)
  const unsigned short* qrow = qbuf + ((size_t)n * LL + m0) * HH;
  for (int c = tid; c < 16 * (HH / 8); c += 256) {
    int row = c >> 7, col8 = c & 127;
    stage16(qrow + row * HH + col8 * 8, &qs[row * QS + col8 * 8]);
  }
  stage_wait_barrier();

  const int w = tid >> 5, lane = tid & 31, half = lane >> 4, l16 = lane & 15;
  const float inv_sqrt_l = 0.022097086912079608f;  // 1/sqrt(2048) (seq-len scale)

  // phase 1: scores S = q @ k^T * inv_sqrt_l (ping-pong B fragment)
  for (int i = 0; i < 16; ++i) {
    const int mt = w + 8 * i;              // m tile 0..127
    v8f acc = {};
    const unsigned short* kcol =
        kbuf + ((size_t)n * LL + mt * 16 + l16) * HH + half * 16;
    __builtin_prefetch(kcol + 8 * 16 * HH, 0, 1);  // next m tile for this wave
    const unsigned short* arow = &qs[l16 * QS];

    v16bf b0 = load_b_frag(kcol);
    v16bf b1;
    int ks = 0;
    #pragma unroll 1
    for (; ks < HH / 32 - 2; ks += 2) {
      b1 = load_b_frag(kcol + (ks + 1) * 32);
      v16bf a = load_a_frag(arow + ks * 32, half);
      acc = wmma_bf16(a, b0, acc);
      b0 = load_b_frag(kcol + (ks + 2) * 32);
      a = load_a_frag(arow + (ks + 1) * 32, half);
      acc = wmma_bf16(a, b1, acc);
    }
    b1 = load_b_frag(kcol + (ks + 1) * 32);
    v16bf a = load_a_frag(arow + ks * 32, half);
    acc = wmma_bf16(a, b0, acc);
    a = load_a_frag(arow + (ks + 1) * 32, half);
    acc = wmma_bf16(a, b1, acc);

    #pragma unroll
    for (int r = 0; r < 8; ++r)
      sc[(r + half * 8) * SS + mt * 16 + l16] = f2bf(acc[r] * inv_sqrt_l);
  }
  __syncthreads();

  // phase 2: softmax (store unnormalized exp; fold 1/rowsum into final store)
  {
    const int row = tid >> 4, sub = tid & 15;
    float lmax = -3.0e38f;
    for (int c2 = sub; c2 < LL; c2 += 16)
      lmax = fmaxf(lmax, bf2f(sc[row * SS + c2]));
    red[row * 16 + sub] = lmax;
    __syncthreads();
    if (sub == 0) {
      float m = red[row * 16];
      for (int i = 1; i < 16; ++i) m = fmaxf(m, red[row * 16 + i]);
      rowmax[row] = m;
    }
    __syncthreads();
    const float rmax = rowmax[row];
    float lsum = 0.f;
    for (int c2 = sub; c2 < LL; c2 += 16) {
      float p = __expf(bf2f(sc[row * SS + c2]) - rmax);
      lsum += p;
      sc[row * SS + c2] = f2bf(p);
    }
    red[row * 16 + sub] = lsum;
    __syncthreads();
    if (sub == 0) {
      float s = 0.f;
      for (int i = 0; i < 16; ++i) s += red[row * 16 + i];
      rowsum[row] = s;
    }
    __syncthreads();
  }

  // phase 3: out = P @ v (8 live accumulators, batched + double-buffered B)
  v8f acc[8];
  {
    const v8f zero = {};
    #pragma unroll
    for (int j = 0; j < 8; ++j) acc[j] = zero;
  }
  const unsigned short* arow = &sc[l16 * SS];
  const unsigned short* vbase =
      vT + ((size_t)n * OO + (w * 8) * 16 + l16) * LL + half * 16;
  const size_t cstride = (size_t)16 * LL;

  {
    v16bf bA[8], bB[8];
    load_b8(bA, vbase, cstride, 0);
    int ks = 0;
    #pragma unroll 1
    for (; ks < LL / 32 - 2; ks += 2) {
      load_b8(bB, vbase, cstride, (ks + 1) * 32);
      v16bf a = load_a_frag(arow + ks * 32, half);
      wmma8(acc, a, bA);
      load_b8(bA, vbase, cstride, (ks + 2) * 32);
      a = load_a_frag(arow + (ks + 1) * 32, half);
      wmma8(acc, a, bB);
    }
    load_b8(bB, vbase, cstride, (ks + 1) * 32);
    v16bf a = load_a_frag(arow + ks * 32, half);
    wmma8(acc, a, bA);
    a = load_a_frag(arow + (ks + 1) * 32, half);
    wmma8(acc, a, bB);
  }

  #pragma unroll
  for (int j = 0; j < 8; ++j) {
    const int o = (w * 8 + j) * 16 + l16;
    #pragma unroll
    for (int r = 0; r < 8; ++r) {
      const int row = r + half * 8;
      out[((size_t)n * LL + m0 + row) * OO + o] = acc[j][r] / rowsum[row];
    }
  }
}

// ---------------- host launcher ----------------
extern "C" void kernel_launch(void* const* d_in, const int* in_sizes, int n_in,
                              void* d_out, int out_size, void* d_ws, size_t ws_size,
                              hipStream_t stream) {
  const float* x  = (const float*)d_in[0];
  const float* Wq = (const float*)d_in[1];
  const float* bq = (const float*)d_in[2];
  const float* Wk = (const float*)d_in[3];
  const float* bk = (const float*)d_in[4];
  const float* Wv = (const float*)d_in[5];
  const float* bv = (const float*)d_in[6];
  float* out = (float*)d_out;

  char* ws = (char*)d_ws;
  const size_t MB = 1024ull * 1024ull;
  unsigned short* xb   = (unsigned short*)(ws + 0);        // 16 MiB
  unsigned short* WqT  = (unsigned short*)(ws + 16 * MB);  //  2 MiB
  unsigned short* WkT  = (unsigned short*)(ws + 18 * MB);  //  2 MiB
  unsigned short* WvT  = (unsigned short*)(ws + 20 * MB);  //  2 MiB
  unsigned short* qbuf = (unsigned short*)(ws + 22 * MB);  // 16 MiB
  unsigned short* kbuf = (unsigned short*)(ws + 38 * MB);  // 16 MiB
  unsigned short* vT   = (unsigned short*)(ws + 54 * MB);  // 16 MiB (70 MiB total)

  cvt_x_kernel<<<(NB * LL * DD / 2 + 255) / 256, 256, 0, stream>>>(
      x, xb, NB * LL * DD / 2);
  cvt_wT_kernel<<<(DD * HH) / 256, 256, 0, stream>>>(Wq, WqT);
  cvt_wT_kernel<<<(DD * HH) / 256, 256, 0, stream>>>(Wk, WkT);
  cvt_wT_kernel<<<(DD * OO) / 256, 256, 0, stream>>>(Wv, WvT);
  qkv_kernel<<<NB * LL / 16, 256, 0, stream>>>(xb, WqT, bq, WkT, bk, WvT, bv,
                                               qbuf, kbuf, vT);
  attn_kernel<<<NB * (LL / 16), 256, 0, stream>>>(qbuf, kbuf, vT, out);
}
